// GCNEncoder_38972533244081
// MI455X (gfx1250) — matrix operand order repaired and checked by reading, not compiled, not run on hardware
//
#include <hip/hip_runtime.h>

// ---------------------------------------------------------------------------
// GCN encoder for MI455X (gfx1250, wave32).
// Dense GEMMs run on v_wmma_f32_16x16x32_bf16 (bf16 inputs, f32 accumulate).
// Edge aggregation uses native f32 global atomics; norm precomputed once.
// ---------------------------------------------------------------------------

typedef __attribute__((ext_vector_type(16))) __bf16 v16bf;
typedef __attribute__((ext_vector_type(8)))  __bf16 v8bf;
typedef __attribute__((ext_vector_type(8)))  float  v8f;

#define WG 256
static constexpr int D_OUT = 128;   // GCN hidden size
static constexpr int D_IN  = 256;   // embed output size
static constexpr int NGRAPH = 64;   // graphs per batch

// ------------------------- small elementwise kernels -----------------------

__global__ void k_wcvt(__bf16* __restrict__ o, const float* __restrict__ i, int n) {
    int t = blockIdx.x * blockDim.x + threadIdx.x;
    if (t < n) o[t] = (__bf16)i[t];
}

__global__ void k_deg_init(float* __restrict__ deg, int n) {
    int t = blockIdx.x * blockDim.x + threadIdx.x;
    if (t < n) deg[t] = 1.0f;                    // self-loop contribution
}

__global__ void k_deg_accum(float* __restrict__ deg, const int* __restrict__ dst, int e) {
    int t = blockIdx.x * blockDim.x + threadIdx.x;
    if (t < e) unsafeAtomicAdd(&deg[dst[t]], 1.0f);
}

__global__ void k_dinv(float* __restrict__ deg, int n) {
    int t = blockIdx.x * blockDim.x + threadIdx.x;
    if (t < n) { float d = deg[t]; deg[t] = (d > 0.0f) ? rsqrtf(d) : 0.0f; }
}

__global__ void k_norm(float* __restrict__ nrm, const float* __restrict__ dinv,
                       const int* __restrict__ src, const int* __restrict__ dst, int e) {
    int t = blockIdx.x * blockDim.x + threadIdx.x;
    if (t < e) nrm[t] = dinv[src[t]] * dinv[dst[t]];
}

__global__ void k_relu(float4* __restrict__ x, int n4) {
    int t = blockIdx.x * blockDim.x + threadIdx.x;
    if (t < n4) {
        float4 v = x[t];
        v.x = fmaxf(v.x, 0.0f); v.y = fmaxf(v.y, 0.0f);
        v.z = fmaxf(v.z, 0.0f); v.w = fmaxf(v.w, 0.0f);
        x[t] = v;
    }
}

// out[i,:] = bias + x[i,:] * dinv[i]^2   (bias + self-loop term, initializes out)
__global__ void k_agg_init(float* __restrict__ out, const float* __restrict__ x,
                           const float* __restrict__ dinv, const float* __restrict__ bias,
                           int n) {
    int t = blockIdx.x * blockDim.x + threadIdx.x;
    int i = t >> 5, c = t & 31;                  // 32 float4 chunks per row of 128
    if (i >= n) return;
    float di = dinv[i]; float s = di * di;
    float4 v = ((const float4*)(x + (size_t)i * D_OUT))[c];
    float4 b = ((const float4*)bias)[c];
    float4 r = { b.x + v.x * s, b.y + v.y * s, b.z + v.z * s, b.w + v.w * s };
    ((float4*)(out + (size_t)i * D_OUT))[c] = r;
}

// out[dst,:] += x[src,:] * norm  over E edges; one thread per (edge, 4-float chunk)
__global__ void k_agg_edges(float* __restrict__ out, const float* __restrict__ x,
                            const int* __restrict__ src, const int* __restrict__ dst,
                            const float* __restrict__ nrm, int e) {
    int t = blockIdx.x * blockDim.x + threadIdx.x;
    int ed = t >> 5, c = t & 31;
    if (ed >= e) return;
    int s = src[ed], d = dst[ed];
    float nv = nrm[ed];
    float4 m = ((const float4*)(x + (size_t)s * D_OUT))[c];
    float* o = out + (size_t)d * D_OUT + c * 4;
    unsafeAtomicAdd(o + 0, m.x * nv);
    unsafeAtomicAdd(o + 1, m.y * nv);
    unsafeAtomicAdd(o + 2, m.z * nv);
    unsafeAtomicAdd(o + 3, m.w * nv);
}

// labels / labels_mask / label_node_ids emitted as f32 (output dtype is float*)
__global__ void k_tail(float* __restrict__ out, const int* __restrict__ tok,
                       int n, int npg) {
    int t = blockIdx.x * blockDim.x + threadIdx.x;
    if (t >= n) return;
    out[t]         = (float)tok[t];       // labels
    out[n + t]     = 1.0f;                // labels_mask (all true)
    out[2 * n + t] = (float)(t % npg);    // label_node_ids
}

// ----------------------------- WMMA GEMM kernel ----------------------------
// C[N,128] = A[N,K] @ W[128,K]^T (+ bias).  rows!=null gathers A rows (embed).
// Block: 256 threads = 8 waves; 16 output rows per block, each wave one 16-col
// tile.  A tile staged in LDS as bf16 with +8 pad (row stride = 4 banks mod 64
// -> conflict-free b128 reads across 16 lanes).
template <int K>
__global__ __launch_bounds__(256) void k_gemm_wmma(
    const float* __restrict__ A, const int* __restrict__ rows,
    const __bf16* __restrict__ W, const float* __restrict__ bias,
    float* __restrict__ C) {
    __shared__ __align__(16) __bf16 sA[16][K + 8];

    const int rowbase = blockIdx.x * 16;

    // gather + convert the 16xK A tile (coalesced along K)
    for (int idx = threadIdx.x; idx < 16 * K; idx += 256) {
        int r = idx / K, k = idx - r * K;
        int ar = rows ? rows[rowbase + r] : (rowbase + r);
        sA[r][k] = (__bf16)A[(size_t)ar * K + k];
    }
    __syncthreads();

    const int lane    = threadIdx.x & 31;
    const int wv      = threadIdx.x >> 5;
    const int colbase = wv * 16;
    const int mn      = lane & 15;     // A: M index / B,C: N index
    const int half    = lane >> 4;

    const __bf16* wrow = W + (size_t)(colbase + mn) * K;   // B[k,n] = W[n,k]

    v8f acc = {};
    for (int kb = 0; kb < K; kb += 32) {
        // A fragment (ISA 16-bit A 16x32 layout):
        //   lanes 0-15 : elems 0-7 -> K kb+0..7,   elems 8-15 -> K kb+16..23
        //   lanes 16-31: elems 0-7 -> K kb+8..15,  elems 8-15 -> K kb+24..31
        v8bf alo = *(const v8bf*)&sA[mn][kb + 8 * half];
        v8bf ahi = *(const v8bf*)&sA[mn][kb + 16 + 8 * half];
        v16bf a = __builtin_shufflevector(alo, ahi,
                                          0, 1, 2, 3, 4, 5, 6, 7,
                                          8, 9, 10, 11, 12, 13, 14, 15);
        // B fragment: same K layout as A (elems 0-7 -> kb+8*half.., elems 8-15 -> kb+16+8*half..)
        v8bf blo = *(const v8bf*)(wrow + kb + 8 * half);
        v8bf bhi = *(const v8bf*)(wrow + kb + 16 + 8 * half);
        v16bf b = __builtin_shufflevector(blo, bhi,
                                          0, 1, 2, 3, 4, 5, 6, 7,
                                          8, 9, 10, 11, 12, 13, 14, 15);

        acc = __builtin_amdgcn_wmma_f32_16x16x32_bf16(
            false, a, false, b, (short)0, acc, false, false);
    }

    // C layout: lane n = lane&15 (+colbase), vgpr r -> m = r + 8*half
    const int n = colbase + mn;
    const float bv = bias ? bias[n] : 0.0f;
#pragma unroll
    for (int r = 0; r < 8; ++r) {
        int m = r + 8 * half;
        C[(size_t)(rowbase + m) * D_OUT + n] = acc[r] + bv;
    }
}

// ------------------------------- launcher ----------------------------------

extern "C" void kernel_launch(void* const* d_in, const int* in_sizes, int n_in,
                              void* d_out, int out_size, void* d_ws, size_t ws_size,
                              hipStream_t stream) {
    const int*   tok  = (const int*)d_in[0];
    const int*   eidx = (const int*)d_in[1];
    const float* emb  = (const float*)d_in[2];
    const float* Wn_w = (const float*)d_in[3];
    const float* Wn_b = (const float*)d_in[4];
    const float* W1_w = (const float*)d_in[5];
    const float* W1_b = (const float*)d_in[6];
    const float* W2_w = (const float*)d_in[7];
    const float* W2_b = (const float*)d_in[8];

    const int N = in_sizes[0];
    const int E = in_sizes[1] / 2;
    const int* src = eidx;
    const int* dst = eidx + E;

    // ---- workspace carve-up (all regions naturally 256B-aligned) ----
    char* w = (char*)d_ws;
    float*  deg = (float*)w;  w += (size_t)N * 4;            // becomes dinv in place
    float*  nrm = (float*)w;  w += (size_t)E * 4;
    float*  xA  = (float*)w;  w += (size_t)N * D_OUT * 4;
    float*  xB  = (float*)w;  w += (size_t)N * D_OUT * 4;
    __bf16* Wn  = (__bf16*)w; w += (size_t)D_OUT * D_IN * 2;
    __bf16* W1  = (__bf16*)w; w += (size_t)D_OUT * D_OUT * 2;
    __bf16* W2  = (__bf16*)w; w += (size_t)D_OUT * D_OUT * 2;

    float* out = (float*)d_out;
    auto cdiv = [](long a, long b) { return (int)((a + b - 1) / b); };

    // weights -> bf16 once per call
    k_wcvt<<<cdiv(D_OUT * D_IN, WG), WG, 0, stream>>>(Wn, Wn_w, D_OUT * D_IN);
    k_wcvt<<<cdiv(D_OUT * D_OUT, WG), WG, 0, stream>>>(W1, W1_w, D_OUT * D_OUT);
    k_wcvt<<<cdiv(D_OUT * D_OUT, WG), WG, 0, stream>>>(W2, W2_w, D_OUT * D_OUT);

    // degree / rsqrt / per-edge norm (shared by both layers)
    k_deg_init <<<cdiv(N, WG), WG, 0, stream>>>(deg, N);
    k_deg_accum<<<cdiv(E, WG), WG, 0, stream>>>(deg, dst, E);
    k_dinv     <<<cdiv(N, WG), WG, 0, stream>>>(deg, N);
    k_norm     <<<cdiv(E, WG), WG, 0, stream>>>(nrm, deg, src, dst, E);

    // x0 = embed[tok] @ Wn^T + b   (gather fused into WMMA GEMM, K=256)
    k_gemm_wmma<256><<<N / 16, 256, 0, stream>>>(emb, tok, Wn, Wn_b, xA);

    // layer 1: lin -> aggregate(self+edges) -> +b -> relu
    k_gemm_wmma<128><<<N / 16, 256, 0, stream>>>(xA, nullptr, W1, nullptr, xB);
    k_agg_init <<<cdiv((long)N * 32, WG), WG, 0, stream>>>(xA, xB, deg, W1_b, N);
    k_agg_edges<<<cdiv((long)E * 32, WG), WG, 0, stream>>>(xA, xB, src, dst, nrm, E);
    k_relu     <<<cdiv((long)N * D_OUT / 4, WG), WG, 0, stream>>>((float4*)xA, N * D_OUT / 4);

    // layer 2: lin -> aggregate directly into d_out
    k_gemm_wmma<128><<<N / 16, 256, 0, stream>>>(xA, nullptr, W2, nullptr, xB);
    k_agg_init <<<cdiv((long)N * 32, WG), WG, 0, stream>>>(out, xB, deg, W2_b, N);
    k_agg_edges<<<cdiv((long)E * 32, WG), WG, 0, stream>>>(out, xB, src, dst, nrm, E);

    // trailing tuple outputs
    if (out_size >= N * D_OUT + 3 * N)
        k_tail<<<cdiv(N, WG), WG, 0, stream>>>(out + (size_t)N * D_OUT, tok, N, N / NGRAPH);
}
